// PointNetSAModule_52518860095859
// MI455X (gfx1250) — matrix operand order repaired and checked by reading, not compiled, not run on hardware
//
#include <hip/hip_runtime.h>

#define N_PTS   8192
#define B_SZ    8
#define M_CTR   1024
#define K_NB    64
#define C_IN    64
#define RAD2    0.04f
#define BN_EPSF 1e-5f

typedef __attribute__((ext_vector_type(16))) __bf16 v16bf;
typedef __attribute__((ext_vector_type(8)))  float  v8f;

#define X1S 104   // LDS column stride (bf16 elems) for 96-row layer-1 input
#define X2S 72    // LDS column stride for 64-row layer-2/3 inputs

__device__ __forceinline__ unsigned short f2bf(float f) {
    union { float f; unsigned u; } x; x.f = f;
    unsigned u = x.u;
    unsigned r = u + 0x7FFFu + ((u >> 16) & 1u);   // round-to-nearest-even
    return (unsigned short)(r >> 16);
}

// ---------------------------------------------------------------------------
// 1) Farthest point sampling: one 1024-thread workgroup per batch.
//    Per-point coords & running min-dist live in registers (8 pts/thread,
//    strided p = t + 1024*i so strictly-greater local compare keeps the
//    smallest index on ties, matching jnp.argmax first-occurrence).
//    Writes centers (B,3,M) directly into the output tail region.
// ---------------------------------------------------------------------------
__global__ __launch_bounds__(1024)
void fps_kernel(const float* __restrict__ coords, float* __restrict__ centers_out)
{
    const int b = blockIdx.x;
    const int t = threadIdx.x;
    const float* cb = coords + (size_t)b * 3 * N_PTS;

    float px[8], py[8], pz[8], dist[8];
#pragma unroll
    for (int i = 0; i < 8; ++i) {
        int p = t + i * 1024;
        px[i] = cb[p];
        py[i] = cb[N_PTS + p];
        pz[i] = cb[2 * N_PTS + p];
        dist[i] = 3.4e38f;
    }

    __shared__ float s_val[1024];
    __shared__ int   s_idx[1024];
    __shared__ float s_c[3];

    float* cout = centers_out + (size_t)b * 3 * M_CTR;
    int far = 0;

    for (int iter = 0; iter < M_CTR; ++iter) {
        if ((far & 1023) == t) {                 // owner publishes centroid
            int i = far >> 10;
            s_c[0] = px[i]; s_c[1] = py[i]; s_c[2] = pz[i];
            cout[iter]             = px[i];
            cout[M_CTR + iter]     = py[i];
            cout[2 * M_CTR + iter] = pz[i];
        }
        __syncthreads();
        float cx = s_c[0], cy = s_c[1], cz = s_c[2];

        float bv = -1.0f; int bi = 0;
#pragma unroll
        for (int i = 0; i < 8; ++i) {
            float dx = px[i] - cx, dy = py[i] - cy, dz = pz[i] - cz;
            float d  = dx * dx + dy * dy + dz * dz;
            dist[i]  = fminf(dist[i], d);
            if (dist[i] > bv) { bv = dist[i]; bi = t + i * 1024; }
        }
        s_val[t] = bv; s_idx[t] = bi;
        __syncthreads();
        for (int off = 512; off > 0; off >>= 1) {
            if (t < off) {
                float v2 = s_val[t + off]; int i2 = s_idx[t + off];
                if (v2 > s_val[t] || (v2 == s_val[t] && i2 < s_idx[t])) {
                    s_val[t] = v2; s_idx[t] = i2;
                }
            }
            __syncthreads();
        }
        far = s_idx[0];
        __syncthreads();
    }
}

// ---------------------------------------------------------------------------
// 2) Prep: fold BN into per-channel scale/shift, convert weights to bf16
//    (W1 zero-padded from K=67 to K=96 so layer 1 is 3 WMMA k-steps).
// ---------------------------------------------------------------------------
__global__ __launch_bounds__(256)
void prep_kernel(const float* __restrict__ W1, const float* __restrict__ W2,
                 const float* __restrict__ W3,
                 const float* b1, const float* b2, const float* b3,
                 const float* g1, const float* g2, const float* g3,
                 const float* be1, const float* be2, const float* be3,
                 const float* mu1, const float* mu2, const float* mu3,
                 const float* va1, const float* va2, const float* va3,
                 unsigned short* __restrict__ W1b, unsigned short* __restrict__ W2b,
                 unsigned short* __restrict__ W3b, float* __restrict__ ss)
{
    const int t = threadIdx.x;
    if (t < 64) {
        float sc = g1[t] * rsqrtf(va1[t] + BN_EPSF);
        ss[t]        = sc;
        ss[64 + t]   = be1[t] + sc * (b1[t] - mu1[t]);
        float sc2 = g2[t] * rsqrtf(va2[t] + BN_EPSF);
        ss[128 + t]  = sc2;
        ss[192 + t]  = be2[t] + sc2 * (b2[t] - mu2[t]);
    }
    if (t < 128) {
        float sc = g3[t] * rsqrtf(va3[t] + BN_EPSF);
        ss[256 + t]  = sc;
        ss[384 + t]  = be3[t] + sc * (b3[t] - mu3[t]);
    }
    for (int e = t; e < 64 * 96; e += 256) {
        int r = e / 96, c = e - r * 96;
        W1b[e] = f2bf(c < 67 ? W1[r * 67 + c] : 0.0f);
    }
    for (int e = t; e < 64 * 64; e += 256)  W2b[e] = f2bf(W2[e]);
    for (int e = t; e < 128 * 64; e += 256) W3b[e] = f2bf(W3[e]);
}

// ---------------------------------------------------------------------------
// 3) Ball query: one wave32 per center; wave-ballot compaction emits the
//    first 64 in-radius indices in ascending order, padded with the first
//    hit (0 if none) — matches the sort/where trick in the reference.
// ---------------------------------------------------------------------------
__global__ __launch_bounds__(256)
void ballq_kernel(const float* __restrict__ coords, const float* __restrict__ centers,
                  int* __restrict__ nb_idx)
{
    const int lane = threadIdx.x & 31;
    const int wv   = threadIdx.x >> 5;
    const int cidx = blockIdx.x * 8 + wv;          // 0..8191
    const int b = cidx >> 10, m = cidx & 1023;

    const float* cb  = coords + (size_t)b * 3 * N_PTS;
    const float* ctr = centers + (size_t)b * 3 * M_CTR;
    const float cx = ctr[m], cy = ctr[M_CTR + m], cz = ctr[2 * M_CTR + m];

    int* out = nb_idx + (size_t)cidx * K_NB;
    int cnt = 0, first_idx = 0;

    for (int base = 0; base < N_PTS; base += 32) {
        int p = base + lane;
        float dx = cb[p] - cx, dy = cb[N_PTS + p] - cy, dz = cb[2 * N_PTS + p] - cz;
        float d2 = dx * dx + dy * dy + dz * dz;
        bool hit = d2 < RAD2;
        unsigned bal = __builtin_amdgcn_ballot_w32(hit);
        if (cnt == 0 && bal) first_idx = base + __builtin_ctz(bal);
        if (hit) {
            int slot = cnt + __builtin_popcount(bal & ((1u << lane) - 1u));
            if (slot < K_NB) out[slot] = p;
        }
        cnt += __builtin_popcount(bal);
        if (cnt >= K_NB) break;
    }
    int pad = cnt < K_NB ? cnt : K_NB;
    for (int s = pad + lane; s < K_NB; s += 32) out[s] = first_idx;
}

// ---------------------------------------------------------------------------
// 4) Gather + 3-layer shared MLP (bf16 WMMA, f32 accum) + max-pool.
//    128 threads (4 waves) per center. X staged column-major bf16 in LDS so
//    B-fragments are two contiguous 8-elem column segments; A-fragments loaded
//    from bf16 weights per the CDNA5 16-bit A layout.
// ---------------------------------------------------------------------------
__global__ __launch_bounds__(128)
void mlp_kernel(const float* __restrict__ feats, const float* __restrict__ coords,
                const float* __restrict__ centers, const int* __restrict__ nb_idx,
                const unsigned short* __restrict__ W1b,
                const unsigned short* __restrict__ W2b,
                const unsigned short* __restrict__ W3b,
                const float* __restrict__ ss, float* __restrict__ out)
{
    __shared__ unsigned short X1[64 * X1S];
    __shared__ unsigned short X2[64 * X2S];
    __shared__ unsigned short X3[64 * X2S];
    __shared__ int   sidx[K_NB];
    __shared__ float sctr[3];

    const int tid  = threadIdx.x;
    const int lane = tid & 31, w = tid >> 5;
    const int h = lane >> 4, ln = lane & 15;
    const int cidx = blockIdx.x;
    const int b = cidx >> 10, m = cidx & 1023;

    if (tid < K_NB) sidx[tid] = nb_idx[(size_t)cidx * K_NB + tid];
    if (tid < 3)    sctr[tid] = centers[(size_t)b * 3 * M_CTR + tid * M_CTR + m];
    __syncthreads();

    // --- gather X1 = [rel coords(3); feats(64); zero pad -> 96] x 64 cols ---
    const float* cb = coords + (size_t)b * 3 * N_PTS;
    const float* fb = feats  + (size_t)b * C_IN * N_PTS;
    for (int e = tid; e < 64 * 96; e += 128) {
        int col = e / 96, c = e - col * 96;
        int p = sidx[col];
        float v;
        if (c < 3)       v = cb[c * N_PTS + p] - sctr[c];
        else if (c < 67) v = fb[(c - 3) * N_PTS + p];
        else             v = 0.0f;
        X1[col * X1S + c] = f2bf(v);
    }
    __syncthreads();

    // A fragment: lane M=ln, K blocks [8h,8h+8) and [16+8h,24+8h) per k-step
    auto loadA = [&](const unsigned short* W, int ldk, int mrow, int kbase) -> v16bf {
        union { v16bf v; unsigned short s[16]; } u;
        const unsigned short* row = W + (size_t)(mrow + ln) * ldk + kbase + 8 * h;
#pragma unroll
        for (int j = 0; j < 16; ++j) u.s[j] = row[(j < 8) ? j : (j + 8)];
        return u.v;
    };
    // B fragment: lane N=16*ntile+ln, K blocks [8h,8h+8) and [16+8h,24+8h) per k-step
    auto loadB = [&](const unsigned short* Xs, int lds_, int ntile, int kbase) -> v16bf {
        union { v16bf v; unsigned short s[16]; } u;
        const unsigned short* colp = Xs + (16 * ntile + ln) * lds_ + kbase + 8 * h;
#pragma unroll
        for (int j = 0; j < 16; ++j) u.s[j] = colp[(j < 8) ? j : (j + 8)];
        return u.v;
    };

    // ----------------- layer 1: (64x96) @ (96x64) ------------------
    {
        v8f acc[4] = {};
        v16bf a0 = loadA(W1b, 96, 16 * w, 0);
        v16bf a1 = loadA(W1b, 96, 16 * w, 32);
        v16bf a2 = loadA(W1b, 96, 16 * w, 64);
#pragma unroll
        for (int n = 0; n < 4; ++n) {
            v16bf bB = loadB(X1, X1S, n, 0);
            acc[n] = __builtin_amdgcn_wmma_f32_16x16x32_bf16(false, a0, false, bB, (short)0, acc[n], false, false);
            bB = loadB(X1, X1S, n, 32);
            acc[n] = __builtin_amdgcn_wmma_f32_16x16x32_bf16(false, a1, false, bB, (short)0, acc[n], false, false);
            bB = loadB(X1, X1S, n, 64);
            acc[n] = __builtin_amdgcn_wmma_f32_16x16x32_bf16(false, a2, false, bB, (short)0, acc[n], false, false);
        }
#pragma unroll
        for (int r = 0; r < 8; ++r) {
            int o = 16 * w + r + 8 * h;
            float sc = ss[o], sh = ss[64 + o];
#pragma unroll
            for (int n = 0; n < 4; ++n) {
                float y = fmaxf(0.0f, sc * acc[n][r] + sh);
                X2[(16 * n + ln) * X2S + o] = f2bf(y);
            }
        }
    }
    __syncthreads();

    // ----------------- layer 2: (64x64) @ (64x64) ------------------
    {
        v8f acc[4] = {};
        v16bf a0 = loadA(W2b, 64, 16 * w, 0);
        v16bf a1 = loadA(W2b, 64, 16 * w, 32);
#pragma unroll
        for (int n = 0; n < 4; ++n) {
            v16bf bB = loadB(X2, X2S, n, 0);
            acc[n] = __builtin_amdgcn_wmma_f32_16x16x32_bf16(false, a0, false, bB, (short)0, acc[n], false, false);
            bB = loadB(X2, X2S, n, 32);
            acc[n] = __builtin_amdgcn_wmma_f32_16x16x32_bf16(false, a1, false, bB, (short)0, acc[n], false, false);
        }
#pragma unroll
        for (int r = 0; r < 8; ++r) {
            int o = 16 * w + r + 8 * h;
            float sc = ss[128 + o], sh = ss[192 + o];
#pragma unroll
            for (int n = 0; n < 4; ++n) {
                float y = fmaxf(0.0f, sc * acc[n][r] + sh);
                X3[(16 * n + ln) * X2S + o] = f2bf(y);
            }
        }
    }
    __syncthreads();

    // ------------- layer 3: (128x64) @ (64x64) + max over K -------------
#pragma unroll
    for (int s = 0; s < 2; ++s) {
        int mrow = 16 * w + 64 * s;                 // strips w and w+4
        v8f acc[4] = {};
        v16bf a0 = loadA(W3b, 64, mrow, 0);
        v16bf a1 = loadA(W3b, 64, mrow, 32);
#pragma unroll
        for (int n = 0; n < 4; ++n) {
            v16bf bB = loadB(X3, X2S, n, 0);
            acc[n] = __builtin_amdgcn_wmma_f32_16x16x32_bf16(false, a0, false, bB, (short)0, acc[n], false, false);
            bB = loadB(X3, X2S, n, 32);
            acc[n] = __builtin_amdgcn_wmma_f32_16x16x32_bf16(false, a1, false, bB, (short)0, acc[n], false, false);
        }
#pragma unroll
        for (int r = 0; r < 8; ++r) {
            int o = mrow + r + 8 * h;
            float sc = ss[256 + o], sh = ss[384 + o];
            float mx = 0.0f;                         // ReLU output >= 0
#pragma unroll
            for (int n = 0; n < 4; ++n)
                mx = fmaxf(mx, fmaxf(0.0f, sc * acc[n][r] + sh));
            // max across the 16 lanes of this half (wave32 xor shuffles)
            mx = fmaxf(mx, __shfl_xor(mx, 1, 32));
            mx = fmaxf(mx, __shfl_xor(mx, 2, 32));
            mx = fmaxf(mx, __shfl_xor(mx, 4, 32));
            mx = fmaxf(mx, __shfl_xor(mx, 8, 32));
            if (ln == 0)
                out[((size_t)b * 128 + o) * M_CTR + m] = mx;
        }
    }
}

// ---------------------------------------------------------------------------
extern "C" void kernel_launch(void* const* d_in, const int* in_sizes, int n_in,
                              void* d_out, int out_size, void* d_ws, size_t ws_size,
                              hipStream_t stream)
{
    (void)in_sizes; (void)n_in; (void)out_size; (void)ws_size;

    const float* feats  = (const float*)d_in[0];
    const float* coords = (const float*)d_in[1];
    const float* W1 = (const float*)d_in[2];
    const float* W2 = (const float*)d_in[3];
    const float* W3 = (const float*)d_in[4];
    const float* b1 = (const float*)d_in[5];
    const float* b2 = (const float*)d_in[6];
    const float* b3 = (const float*)d_in[7];
    const float* g1 = (const float*)d_in[8];
    const float* g2 = (const float*)d_in[9];
    const float* g3 = (const float*)d_in[10];
    const float* be1 = (const float*)d_in[11];
    const float* be2 = (const float*)d_in[12];
    const float* be3 = (const float*)d_in[13];
    const float* mu1 = (const float*)d_in[14];
    const float* mu2 = (const float*)d_in[15];
    const float* mu3 = (const float*)d_in[16];
    const float* va1 = (const float*)d_in[17];
    const float* va2 = (const float*)d_in[18];
    const float* va3 = (const float*)d_in[19];

    float* out = (float*)d_out;
    float* centers = out + (size_t)B_SZ * 128 * M_CTR;   // output tuple tail

    char* ws = (char*)d_ws;
    int* nb = (int*)ws;                                          // 2 MB
    unsigned short* W1b = (unsigned short*)(ws + 2097152);       // 64x96 bf16
    unsigned short* W2b = (unsigned short*)(ws + 2097152 + 12288);
    unsigned short* W3b = (unsigned short*)(ws + 2097152 + 12288 + 8192);
    float* ss = (float*)(ws + 2097152 + 12288 + 8192 + 16384);   // 512 floats

    fps_kernel  <<<B_SZ, 1024, 0, stream>>>(coords, centers);
    prep_kernel <<<1, 256, 0, stream>>>(W1, W2, W3, b1, b2, b3, g1, g2, g3,
                                        be1, be2, be3, mu1, mu2, mu3,
                                        va1, va2, va3, W1b, W2b, W3b, ss);
    ballq_kernel<<<M_CTR * B_SZ / 8, 256, 0, stream>>>(coords, centers, nb);
    mlp_kernel  <<<B_SZ * M_CTR, 128, 0, stream>>>(feats, coords, centers, nb,
                                                   W1b, W2b, W3b, ss, out);
}
